// GGNN_pr_25220047962222
// MI455X (gfx1250) — matrix operand order repaired and checked by reading, not compiled, not run on hardware
//
#include <hip/hip_runtime.h>
#include <hip/hip_bf16.h>

#define N_NODES  100000
#define N_EDGES  1600000
#define N_GRAPHS 128
#define IN_DIM   64
#define HID      128
#define OUT_DIM  32
#define DOC_DIM  768
#define LAYERS   3

typedef _Float16 v4h  __attribute__((ext_vector_type(4)));
typedef _Float16 v8h  __attribute__((ext_vector_type(8)));
typedef _Float16 v16h __attribute__((ext_vector_type(16)));
typedef float    v8f  __attribute__((ext_vector_type(8)));

// A-fragment (16x32 f16, MxK): lane<16 holds row=lane, K {0..7,16..23};
// lane>=16 holds row=lane-16, K {8..15,24..31}. p points at the first 8-half
// chunk; second chunk is +16 halfs (+32B).
__device__ __forceinline__ v16h afrag(const _Float16* p) {
    union { v16h v; v8h h[2]; } u;
    u.h[0] = *(const v8h*)(p);
    u.h[1] = *(const v8h*)(p + 16);
    return u.v;
}

__device__ __forceinline__ v8f wmma_f16(v16h a, v16h b, v8f c) {
    return __builtin_amdgcn_wmma_f32_16x16x32_f16(
        /*neg_a=*/false, a, /*neg_b=*/false, b,
        /*c_mod=*/(short)0, c, /*reuse_a=*/false, /*reuse_b=*/false);
}

__device__ __forceinline__ float sigmoidf_(float x) {
    return 1.0f / (1.0f + __expf(-x));
}

// Async byte-copy of a 4KB tile (16 rows x 128 halfs) global->LDS.
// Each of the 256 threads issues one 16B GLOBAL_LOAD_ASYNC_TO_LDS_B128
// (GVS mode: saddr base + per-lane vgpr offset). Tracked with ASYNCcnt.
__device__ __forceinline__ void async_tile4k(const void* gbase, uint32_t lds_base) {
    uint32_t off = threadIdx.x * 16;
    asm volatile("global_load_async_to_lds_b128 %0, %1, %2"
                 :: "v"(lds_base + off), "v"(off),
                    "s"((unsigned long long)(uintptr_t)gbase)
                 : "memory");
}
__device__ __forceinline__ void wait_async0() {
    asm volatile("s_wait_asynccnt 0" ::: "memory");
}

// out[n*K + k] = (f16) in[k*N + n]   (weight transpose+convert, [in,out]->[N][K])
__global__ void k_cvt_t(const float* __restrict__ in, _Float16* __restrict__ out,
                        int K, int N) {
    int idx = blockIdx.x * 256 + threadIdx.x;
    if (idx >= K * N) return;
    int n = idx / K, k = idx - n * K;
    out[(size_t)n * K + k] = (_Float16)in[(size_t)k * N + n];
}

// C[16 x 128] tile per block. A: f32 [M,K] (K multiple of 32), BT: f16 [128][K].
// 8 waves; wave w owns output col-tile w. A staged f32->f16 in LDS.
// Stores f32 C and optionally an f16 mirror C16.
__global__ void __launch_bounds__(256) k_gemm_n128(
    const float* __restrict__ A, const _Float16* __restrict__ BT,
    const float* __restrict__ bias, float* __restrict__ C,
    _Float16* __restrict__ C16, int K, int relu) {
    extern __shared__ _Float16 sA[];      // 16*K halfs
    const int rowBase = blockIdx.x * 16;
    const float* Ab = A + (size_t)rowBase * K;
    for (int i = threadIdx.x; i < 16 * K; i += 256) sA[i] = (_Float16)Ab[i];
    __syncthreads();

    const int wave = threadIdx.x >> 5;
    const int lane = threadIdx.x & 31;
    const int r    = lane & 15;
    const int hi   = lane >> 4;
    v8f acc = {};
    const _Float16* bRow = BT + (size_t)(wave * 16 + r) * K;
    for (int kb = 0; kb < K; kb += 32) {
        v16h a = afrag(sA + r * K + kb + hi * 8);
        v16h b = afrag(bRow + kb + hi * 8);
        acc = wmma_f16(a, b, acc);
    }
    const int col = wave * 16 + r;
    const float bb = bias ? bias[col] : 0.0f;
    const size_t base = (size_t)(rowBase + hi * 8) * HID + col;
#pragma unroll
    for (int i = 0; i < 8; ++i) {
        float v = acc[i] + bb;
        if (relu) v = fmaxf(v, 0.0f);
        C[base + (size_t)i * HID] = v;
        if (C16) C16[base + (size_t)i * HID] = (_Float16)v;
    }
}

// m16 = h16 @ ggnn_w[l] : A already f16 in memory -> async byte-copy into LDS,
// output stored f16 (feeds the edge gather).
__global__ void __launch_bounds__(256) k_gemm_h16(
    const _Float16* __restrict__ A16, const _Float16* __restrict__ BT,
    _Float16* __restrict__ C16) {
    __shared__ _Float16 sA[16 * HID];
    const int rowBase = blockIdx.x * 16;
    async_tile4k(A16 + (size_t)rowBase * HID, (uint32_t)(uintptr_t)sA);
    wait_async0();
    __syncthreads();

    const int wave = threadIdx.x >> 5;
    const int lane = threadIdx.x & 31;
    const int r = lane & 15, hi = lane >> 4;
    v8f acc = {};
    const _Float16* bRow = BT + (size_t)(wave * 16 + r) * HID;
    for (int kb = 0; kb < HID; kb += 32) {
        v16h a = afrag(sA + r * HID + kb + hi * 8);
        v16h b = afrag(bRow + kb + hi * 8);
        acc = wmma_f16(a, b, acc);
    }
    const int col = wave * 16 + r;
    _Float16* Cb = C16 + (size_t)(rowBase + hi * 8) * HID + col;
#pragma unroll
    for (int i = 0; i < 8; ++i) Cb[(size_t)i * HID] = (_Float16)acc[i];
}

// agg[dst] += m16[src] (f32 accumulate), 4 columns per thread
__global__ void k_scatter(const _Float16* __restrict__ m16,
                          const int* __restrict__ eidx,
                          float* __restrict__ agg) {
    int t = blockIdx.x * 256 + threadIdx.x;     // < N_EDGES*32
    int e = t >> 5;
    int c = (t & 31) * 4;
    int src = eidx[e];
    int dst = eidx[N_EDGES + e];
    const v4h v = *(const v4h*)(m16 + (size_t)src * HID + c);
    float* a = agg + (size_t)dst * HID + c;
    atomicAdd(a + 0, (float)v[0]); atomicAdd(a + 1, (float)v[1]);
    atomicAdd(a + 2, (float)v[2]); atomicAdd(a + 3, (float)v[3]);
}

// Fused GRU: per block 16 nodes; gi = agg@w_ih, gh = h@w_hh (both K=128,N=384)
// via WMMA, then elementwise gate math in C/D layout; h32/h16 updated in place.
__global__ void __launch_bounds__(256) k_gru(
    const float* __restrict__ agg, float* __restrict__ h32,
    _Float16* __restrict__ h16,
    const _Float16* __restrict__ ihT, const _Float16* __restrict__ hhT,
    const float* __restrict__ b_ih, const float* __restrict__ b_hh) {
    __shared__ _Float16 sA[16 * HID];
    __shared__ _Float16 sH[16 * HID];
    const int rowBase = blockIdx.x * 16;
    // h tile: pure byte copy, overlap with agg conversion below
    async_tile4k(h16 + (size_t)rowBase * HID, (uint32_t)(uintptr_t)sH);
    const float* Ab = agg + (size_t)rowBase * HID;
    for (int i = threadIdx.x; i < 16 * HID; i += 256) sA[i] = (_Float16)Ab[i];
    wait_async0();
    __syncthreads();

    const int wave = threadIdx.x >> 5;
    const int lane = threadIdx.x & 31;
    const int r = lane & 15, hi = lane >> 4;
    v8f gi0 = {}, gi1 = {}, gi2 = {}, gh0 = {}, gh1 = {}, gh2 = {};
    const size_t gstep = (size_t)HID * HID;    // 128 rows of BT per gate
    for (int kb = 0; kb < HID; kb += 32) {
        v16h aA = afrag(sA + r * HID + kb + hi * 8);
        v16h aH = afrag(sH + r * HID + kb + hi * 8);
        const size_t ro = (size_t)(wave * 16 + r) * HID + kb + hi * 8;
        gi0 = wmma_f16(aA, afrag(ihT + ro),             gi0);
        gi1 = wmma_f16(aA, afrag(ihT + ro + gstep),     gi1);
        gi2 = wmma_f16(aA, afrag(ihT + ro + 2 * gstep), gi2);
        gh0 = wmma_f16(aH, afrag(hhT + ro),             gh0);
        gh1 = wmma_f16(aH, afrag(hhT + ro + gstep),     gh1);
        gh2 = wmma_f16(aH, afrag(hhT + ro + 2 * gstep), gh2);
    }
    const int col = wave * 16 + r;
    const float bir = b_ih[col], biz = b_ih[col + HID], bin = b_ih[col + 2 * HID];
    const float bhr = b_hh[col], bhz = b_hh[col + HID], bhn = b_hh[col + 2 * HID];
    const size_t base = (size_t)(rowBase + hi * 8) * HID + col;
#pragma unroll
    for (int i = 0; i < 8; ++i) {
        float ir = gi0[i] + bir, iz = gi1[i] + biz, in_ = gi2[i] + bin;
        float hr = gh0[i] + bhr, hz = gh1[i] + bhz, hn  = gh2[i] + bhn;
        float rr = sigmoidf_(ir + hr);
        float zz = sigmoidf_(iz + hz);
        float nn = tanhf(in_ + rr * hn);
        float ho = h32[base + (size_t)i * HID];
        float hv = (1.0f - zz) * nn + zz * ho;
        h32[base + (size_t)i * HID] = hv;
        h16[base + (size_t)i * HID] = (_Float16)hv;
    }
}

// per-graph sums + counts
__global__ void k_pool(const float* __restrict__ h, const int* __restrict__ batch,
                       float* __restrict__ sums, float* __restrict__ cnt) {
    int t = blockIdx.x * 256 + threadIdx.x;     // < N_NODES*32
    int n = t >> 5;
    int c = (t & 31) * 4;
    int g = batch[n];
    const float4 v = *(const float4*)(h + (size_t)n * HID + c);
    float* s = sums + (size_t)g * HID + c;
    atomicAdd(s + 0, v.x); atomicAdd(s + 1, v.y);
    atomicAdd(s + 2, v.z); atomicAdd(s + 3, v.w);
    if (c == 0) atomicAdd(cnt + g, 1.0f);
}

// one block per graph: mean-pool finalize, concat, LayerNorm, fusion GEMV, heads
__global__ void __launch_bounds__(256) k_fusion(
    const float* __restrict__ sums, const float* __restrict__ cnt,
    const float* __restrict__ doce,
    const float* __restrict__ ln_g, const float* __restrict__ ln_b,
    const float* __restrict__ fus_w, const float* __restrict__ fus_b,
    const float* __restrict__ task_w, const float* __restrict__ task_b,
    const float* __restrict__ time_w, const float* __restrict__ time_b,
    float* __restrict__ out) {
    __shared__ float f[2 * HID];
    __shared__ float red[2 * HID];
    __shared__ float fz[HID];
    const int g = blockIdx.x, t = threadIdx.x;
    float val;
    if (t < HID) {
        float c = fmaxf(cnt[g], 1.0f);
        val = sums[(size_t)g * HID + t] / c;
    } else {
        val = doce[(size_t)g * HID + (t - HID)];
    }
    f[t] = val;
    red[t] = val;
    __syncthreads();
    for (int s = HID; s > 0; s >>= 1) { if (t < s) red[t] += red[t + s]; __syncthreads(); }
    float mu = red[0] / (2.0f * HID);
    __syncthreads();
    float d = val - mu;
    red[t] = d * d;
    __syncthreads();
    for (int s = HID; s > 0; s >>= 1) { if (t < s) red[t] += red[t + s]; __syncthreads(); }
    float var = red[0] / (2.0f * HID);
    float nv = d * rsqrtf(var + 1e-5f) * ln_g[t] + ln_b[t];
    __syncthreads();
    f[t] = nv;
    __syncthreads();
    if (t < HID) {
        float acc = fus_b[t];
        for (int j = 0; j < 2 * HID; ++j) acc += f[j] * fus_w[(size_t)j * HID + t];
        fz[t] = fmaxf(acc, 0.0f);
    }
    __syncthreads();
    if (t < OUT_DIM) {
        float acc = task_b[t];
        for (int j = 0; j < HID; ++j) acc += fz[j] * task_w[(size_t)j * OUT_DIM + t];
        out[(size_t)g * OUT_DIM + t] = acc;
    } else if (t == OUT_DIM) {
        float acc = time_b[0];
        for (int j = 0; j < HID; ++j) acc += fz[j] * time_w[j];
        out[(size_t)N_GRAPHS * OUT_DIM + g] = acc;
    }
}

extern "C" void kernel_launch(void* const* d_in, const int* in_sizes, int n_in,
                              void* d_out, int out_size, void* d_ws, size_t ws_size,
                              hipStream_t stream) {
    const float* x      = (const float*)d_in[0];
    const float* docf   = (const float*)d_in[1];
    const int*   eidx   = (const int*)d_in[2];
    const int*   batch  = (const int*)d_in[3];
    const float* node_w = (const float*)d_in[4];
    const float* node_b = (const float*)d_in[5];
    const float* ggnn_w = (const float*)d_in[6];
    const float* w_ih   = (const float*)d_in[7];
    const float* b_ih   = (const float*)d_in[8];
    const float* w_hh   = (const float*)d_in[9];
    const float* b_hh   = (const float*)d_in[10];
    const float* doc_w  = (const float*)d_in[11];
    const float* doc_b  = (const float*)d_in[12];
    const float* ln_g   = (const float*)d_in[13];
    const float* ln_b   = (const float*)d_in[14];
    const float* fus_w  = (const float*)d_in[15];
    const float* fus_b  = (const float*)d_in[16];
    const float* task_w = (const float*)d_in[17];
    const float* task_b = (const float*)d_in[18];
    const float* time_w = (const float*)d_in[19];
    const float* time_b = (const float*)d_in[20];
    float* out = (float*)d_out;
    (void)in_sizes; (void)n_in; (void)out_size; (void)ws_size;

    char* ws = (char*)d_ws;
    size_t off = 0;
    auto wsalloc = [&](size_t bytes) -> char* {
        char* p = ws + off;
        off = (off + bytes + 255) & ~(size_t)255;
        return p;
    };
    float*    h32  = (float*)   wsalloc((size_t)N_NODES * HID * 4);
    float*    agg  = (float*)   wsalloc((size_t)N_NODES * HID * 4);
    _Float16* h16  = (_Float16*)wsalloc((size_t)N_NODES * HID * 2);
    _Float16* m16  = (_Float16*)wsalloc((size_t)N_NODES * HID * 2);
    float*    sums = (float*)   wsalloc((size_t)N_GRAPHS * HID * 4);
    float*    cntb = (float*)   wsalloc((size_t)N_GRAPHS * 4);
    float*    doce = (float*)   wsalloc((size_t)N_GRAPHS * HID * 4);
    _Float16* node_wT = (_Float16*)wsalloc((size_t)IN_DIM * HID * 2);
    _Float16* ggnn_wT = (_Float16*)wsalloc((size_t)LAYERS * HID * HID * 2);
    _Float16* ihT     = (_Float16*)wsalloc((size_t)HID * 3 * HID * 2);
    _Float16* hhT     = (_Float16*)wsalloc((size_t)HID * 3 * HID * 2);
    _Float16* doc_wT  = (_Float16*)wsalloc((size_t)DOC_DIM * HID * 2);

    // weight transpose+convert to f16 [N][K]
    k_cvt_t<<<(IN_DIM * HID + 255) / 256, 256, 0, stream>>>(node_w, node_wT, IN_DIM, HID);
    for (int l = 0; l < LAYERS; ++l)
        k_cvt_t<<<(HID * HID + 255) / 256, 256, 0, stream>>>(
            ggnn_w + (size_t)l * HID * HID, ggnn_wT + (size_t)l * HID * HID, HID, HID);
    k_cvt_t<<<(HID * 3 * HID + 255) / 256, 256, 0, stream>>>(w_ih, ihT, HID, 3 * HID);
    k_cvt_t<<<(HID * 3 * HID + 255) / 256, 256, 0, stream>>>(w_hh, hhT, HID, 3 * HID);
    k_cvt_t<<<(DOC_DIM * HID + 255) / 256, 256, 0, stream>>>(doc_w, doc_wT, DOC_DIM, HID);

    // node projection + ReLU: h = relu(x @ node_w + node_b) -> f32 + f16 mirror
    k_gemm_n128<<<N_NODES / 16, 256, 16 * IN_DIM * 2, stream>>>(
        x, node_wT, node_b, h32, h16, IN_DIM, 1);

    for (int l = 0; l < LAYERS; ++l) {
        k_gemm_h16<<<N_NODES / 16, 256, 0, stream>>>(
            h16, ggnn_wT + (size_t)l * HID * HID, m16);
        hipMemsetAsync(agg, 0, (size_t)N_NODES * HID * 4, stream);
        k_scatter<<<(N_EDGES * 32) / 256, 256, 0, stream>>>(m16, eidx, agg);
        k_gru<<<N_NODES / 16, 256, 0, stream>>>(agg, h32, h16, ihT, hhT, b_ih, b_hh);
    }

    hipMemsetAsync(sums, 0, (size_t)N_GRAPHS * HID * 4, stream);
    hipMemsetAsync(cntb, 0, (size_t)N_GRAPHS * 4, stream);
    k_pool<<<(N_NODES * 32) / 256, 256, 0, stream>>>(h32, batch, sums, cntb);

    // doc projection + ReLU (f32 only)
    k_gemm_n128<<<N_GRAPHS / 16, 256, 16 * DOC_DIM * 2, stream>>>(
        docf, doc_wT, doc_b, doce, nullptr, DOC_DIM, 1);

    k_fusion<<<N_GRAPHS, 256, 0, stream>>>(sums, cntb, doce, ln_g, ln_b,
                                           fus_w, fus_b, task_w, task_b,
                                           time_w, time_b, out);
}
